// Pairformer_75350906241815
// MI455X (gfx1250) — matrix operand order repaired and checked by reading, not compiled, not run on hardware
//
#include <hip/hip_runtime.h>
#include <hip/hip_bf16.h>

// ---------------------------------------------------------------------------
// Pairformer block for MI455X (gfx1250, wave32, WMMA).
// All GEMMs run through one generic strided/batched bf16 WMMA kernel
// (v_wmma_f32_16x16x32_bf16, f32 accumulate); 64-wide K stages with
// vectorized b128 global->LDS staging on the fast paths; activations carried
// in bf16 between GEMMs; LayerNorm/softmax as wave32 shuffle reductions.
// ---------------------------------------------------------------------------

typedef __attribute__((ext_vector_type(16))) __bf16 v16bf;
typedef __attribute__((ext_vector_type(8)))  float  v8f;

#define NTOK 256
#define CZ   128
#define CS   384
#define NPAIR (NTOK * NTOK)   // 65536
#define HTRI 4
#define DTRI 32
#define HS   16
#define D_S  24

// ---------------------------------------------------------------------------
// Generic strided batched GEMM:  C[b][m,n] = A[b][m,k] * B[b][k,n] (+bias[n])
// A,B bf16; C f32 (optionally mirrored to bf16).  64x64 output tile per
// 128-thread block; 4 waves each own a 16-row stripe and 4 accumulators.
// K advances 64 per LDS stage -> 8 WMMAs per barrier pair.
// Staging fast paths (full tile in-bounds):
//   A: la_c==1       -> 4x int4 loads per thread (32 contiguous bf16)
//   B: lb_r==1       -> same, stored transposed [n][k]
//   B: lb_c==1       -> vector load along n, scalar scatter into [n][k]
// LDS rows padded to 72 elems (144B pitch) for alignment + conflict spread.
// ---------------------------------------------------------------------------
__global__ __launch_bounds__(128)
void gemm_bf16_wmma_kernel(const __bf16* __restrict__ A, long la_r, long la_c, long a_b1, long a_b2,
                           const __bf16* __restrict__ B, long lb_r, long lb_c, long b_b1, long b_b2,
                           float* __restrict__ C, long lc_r, long lc_c, long c_b1, long c_b2,
                           const float* __restrict__ bias, __bf16* __restrict__ Cbf,
                           long M, long N, long K, int B2cnt)
{
    __shared__ alignas(16) __bf16 sA[64][72];
    __shared__ alignas(16) __bf16 sB[64][72];

    const int tid  = threadIdx.x;
    const int wave = tid >> 5;
    const int lane = tid & 31;
    const int msub = lane & 15;
    const int kh   = lane >> 4;

    const long tileM = (long)blockIdx.y * 64;
    const long tileN = (long)blockIdx.x * 64;
    const int  bz = blockIdx.z;
    const int  b1 = bz / B2cnt;
    const int  b2 = bz - b1 * B2cnt;

    const __bf16* Ab = A + (long)b1 * a_b1 + (long)b2 * a_b2;
    const __bf16* Bb = B + (long)b1 * b_b1 + (long)b2 * b_b2;
    const long coff  = (long)b1 * c_b1 + (long)b2 * c_b2;

    const bool mFull = (tileM + 64 <= M);
    const bool nFull = (tileN + 64 <= N);

    v8f acc[4];
#pragma unroll
    for (int t = 0; t < 4; ++t)
#pragma unroll
        for (int r = 0; r < 8; ++r) acc[t][r] = 0.0f;

    const int lrow = tid >> 1;          // 0..63
    const int lks  = (tid & 1) << 5;    // 0 or 32

    for (long k0 = 0; k0 < K; k0 += 64) {
        const bool kFull = (k0 + 64 <= K);

        // ---- stage A tile [64 rows][64 k] ----
        if (mFull && kFull && la_c == 1) {
            const __bf16* src = Ab + (tileM + lrow) * la_r + k0 + lks;
            const int4* s4 = (const int4*)src;
            int4* d4 = (int4*)&sA[lrow][lks];
            d4[0] = s4[0]; d4[1] = s4[1]; d4[2] = s4[2]; d4[3] = s4[3];
            if (k0 + 64 < K) __builtin_prefetch(src + 64, 0, 1);   // global_prefetch_b8
        } else {
            const long gm = tileM + lrow;
#pragma unroll
            for (int kk = 0; kk < 32; ++kk) {
                const long gk = k0 + lks + kk;
                __bf16 v = (__bf16)0.0f;
                if (gm < M && gk < K) v = Ab[gm * la_r + gk * la_c];
                sA[lrow][lks + kk] = v;
            }
        }
        // ---- stage B tile, transposed into [n][k] ----
        if (nFull && kFull && lb_r == 1) {
            const __bf16* src = Bb + (tileN + lrow) * lb_c + k0 + lks;
            const int4* s4 = (const int4*)src;
            int4* d4 = (int4*)&sB[lrow][lks];
            d4[0] = s4[0]; d4[1] = s4[1]; d4[2] = s4[2]; d4[3] = s4[3];
            if (k0 + 64 < K) __builtin_prefetch(src + 64, 0, 1);
        } else if (nFull && kFull && lb_c == 1) {
            const int kidx = lrow;                 // 0..63
            const int nseg = lks;                  // 0 or 32
            const __bf16* src = Bb + (k0 + kidx) * lb_r + tileN + nseg;
            const int4* s4 = (const int4*)src;
            int4 t4[4] = { s4[0], s4[1], s4[2], s4[3] };
            const __bf16* tv = (const __bf16*)t4;
#pragma unroll
            for (int e = 0; e < 32; ++e) sB[nseg + e][kidx] = tv[e];
        } else {
            const long gn = tileN + lrow;
#pragma unroll
            for (int kk = 0; kk < 32; ++kk) {
                const long gk = k0 + lks + kk;
                __bf16 v = (__bf16)0.0f;
                if (gn < N && gk < K) v = Bb[gk * lb_r + gn * lb_c];
                sB[lrow][lks + kk] = v;
            }
        }
        __syncthreads();

#pragma unroll
        for (int ks = 0; ks < 64; ks += 32) {
            // A fragment 16x32 (ISA 7.12.2): lanes 0-15 K=0..7/16..23,
            // lanes 16-31 K=8..15/24..31
            v16bf afrag;
            {
                const int mr = wave * 16 + msub;
#pragma unroll
                for (int vv = 0; vv < 8; ++vv) {
                    const int kb = ((vv < 4) ? (vv * 2) : (16 + (vv - 4) * 2)) + kh * 8 + ks;
                    afrag[2 * vv]     = sA[mr][kb];
                    afrag[2 * vv + 1] = sA[mr][kb + 1];
                }
            }
#pragma unroll
            for (int t = 0; t < 4; ++t) {
                // B fragment 32x16: lane owns column n=lane&15, K=kh*8+0..7/16+kh*8+0..7
                v16bf bfrag;
                const int nr = t * 16 + msub;
#pragma unroll
                for (int e = 0; e < 16; ++e) bfrag[e] = sB[nr][ks + kh * 8 + ((e < 8) ? e : (e + 8))];
                acc[t] = __builtin_amdgcn_wmma_f32_16x16x32_bf16(
                    false, afrag, false, bfrag, (short)0, acc[t], false, false);
            }
        }
        __syncthreads();
    }

    // C/D layout: VGPR r -> row r + kh*8 ; column = lane&15
#pragma unroll
    for (int t = 0; t < 4; ++t) {
        const long ncol = tileN + t * 16 + msub;
        if (ncol >= N) continue;
        const float bv = bias ? bias[ncol] : 0.0f;
#pragma unroll
        for (int r = 0; r < 8; ++r) {
            const long row = tileM + wave * 16 + kh * 8 + r;
            if (row >= M) continue;
            const float v = acc[t][r] + bv;
            const long idx = coff + row * lc_r + ncol * lc_c;
            C[idx] = v;
            if (Cbf) Cbf[idx] = (__bf16)v;
        }
    }
}

// ---------------------------------------------------------------------------
// Wave32 LayerNorm: one wave per row, shuffle reductions, bf16 output.
// ---------------------------------------------------------------------------
__global__ void ln_bf16_kernel(const float* __restrict__ X, const float* __restrict__ g,
                               const float* __restrict__ b, __bf16* __restrict__ Y,
                               long rows, int C)
{
    const long wid = (long)blockIdx.x * (blockDim.x >> 5) + (threadIdx.x >> 5);
    const int lane = threadIdx.x & 31;
    if (wid >= rows) return;
    const float* x = X + wid * C;
    float sum = 0.f, sq = 0.f;
    for (int c = lane; c < C; c += 32) { float v = x[c]; sum += v; sq += v * v; }
#pragma unroll
    for (int off = 16; off; off >>= 1) {
        sum += __shfl_xor(sum, off, 32);
        sq  += __shfl_xor(sq,  off, 32);
    }
    const float mu  = sum / (float)C;
    const float var = sq / (float)C - mu * mu;
    const float rs  = rsqrtf(var + 1e-5f);
    __bf16* y = Y + wid * C;
    for (int c = lane; c < C; c += 32)
        y[c] = (__bf16)((x[c] - mu) * rs * g[c] + b[c]);
}

// act(G)*L -> bf16 ; silu==0: sigmoid gate, silu==1: SiLU
__global__ void glu_bf16_kernel(const float* __restrict__ G, const float* __restrict__ L,
                                __bf16* __restrict__ Y, long n, int silu)
{
    const long i = (long)blockIdx.x * blockDim.x + threadIdx.x;
    if (i >= n) return;
    const float g = G[i];
    const float s = 1.0f / (1.0f + __expf(-g));
    Y[i] = (__bf16)((silu ? g * s : s) * L[i]);
}

__global__ void resgate_kernel(float* __restrict__ Z, const float* __restrict__ G,
                               const float* __restrict__ P, long n)
{
    const long i = (long)blockIdx.x * blockDim.x + threadIdx.x;
    if (i >= n) return;
    Z[i] += (1.0f / (1.0f + __expf(-G[i]))) * P[i];
}

__global__ void add_inplace_kernel(float* __restrict__ Z, const float* __restrict__ D, long n)
{
    const long i = (long)blockIdx.x * blockDim.x + threadIdx.x;
    if (i < n) Z[i] += D[i];
}

__global__ void transpose_pair_kernel(const float* __restrict__ X, float* __restrict__ Y,
                                      int n, int C)
{
    const long idx = (long)blockIdx.x * blockDim.x + threadIdx.x;
    const long total = (long)n * n * C;
    if (idx >= total) return;
    const int  c  = (int)(idx % C);
    const long ij = idx / C;
    const int  j  = (int)(ij % n);
    const int  i  = (int)(ij / n);
    Y[((long)j * n + i) * C + c] = X[idx];
}

__global__ void add_transposed_kernel(float* __restrict__ Z, const float* __restrict__ D,
                                      int n, int C)
{
    const long idx = (long)blockIdx.x * blockDim.x + threadIdx.x;
    const long total = (long)n * n * C;
    if (idx >= total) return;
    const int  c  = (int)(idx % C);
    const long ij = idx / C;
    const int  j  = (int)(ij % n);
    const int  i  = (int)(ij / n);
    Z[idx] += D[((long)j * n + i) * C + c];
}

__global__ void f32_to_bf16_kernel(const float* __restrict__ X, __bf16* __restrict__ Y, long n)
{
    const long i = (long)blockIdx.x * blockDim.x + threadIdx.x;
    if (i < n) Y[i] = (__bf16)X[i];
}

// bf16 [ (p*n+q)*C + c ]  ->  channel-major  Y[c][p][q]  (swap: Y[c][q][p]).
// Puts the triangle-einsum contraction dim at stride 1 for both operands.
__global__ void pair_to_cmajor_bf16_kernel(const __bf16* __restrict__ X,
                                           __bf16* __restrict__ Y,
                                           int n, int C, int swap)
{
    const long idx = (long)blockIdx.x * blockDim.x + threadIdx.x;
    const long total = (long)n * n * C;
    if (idx >= total) return;
    const int  c  = (int)(idx % C);
    const long pq = idx / C;
    const int  q  = (int)(pq % n);
    const int  p  = (int)(pq / n);
    const long dst = (long)c * n * n + (swap ? ((long)q * n + p) : ((long)p * n + q));
    Y[dst] = X[idx];
}

// Triangle-attention softmax: one wave per (i,h,j) row of 256 scores.
// S [i,h,j,k] f32 ; bias B0 [j,k,h] ; P [i,h,j,k] bf16.  Assumes n==256.
__global__ void softmax_tri_kernel(const float* __restrict__ S, const float* __restrict__ B0,
                                   __bf16* __restrict__ P, float scale, int n, int H)
{
    const long wid = (long)blockIdx.x * (blockDim.x >> 5) + (threadIdx.x >> 5);
    const int lane = threadIdx.x & 31;
    const long total = (long)n * H * n;
    if (wid >= total) return;
    const int j = (int)(wid % n);
    const int h = (int)((wid / n) % H);
    const int i = (int)(wid / ((long)n * H));
    const float* srow = S + (((long)i * H + h) * n + j) * (long)n;
    float vals[8];
    float mx = -1e30f;
#pragma unroll
    for (int t = 0; t < 8; ++t) {
        const int k = t * 32 + lane;
        const float v = srow[k] * scale + B0[((long)j * n + k) * H + h];
        vals[t] = v; mx = fmaxf(mx, v);
    }
#pragma unroll
    for (int off = 16; off; off >>= 1) mx = fmaxf(mx, __shfl_xor(mx, off, 32));
    float sum = 0.f;
#pragma unroll
    for (int t = 0; t < 8; ++t) { vals[t] = __expf(vals[t] - mx); sum += vals[t]; }
#pragma unroll
    for (int off = 16; off; off >>= 1) sum += __shfl_xor(sum, off, 32);
    const float inv = 1.0f / sum;
    __bf16* prow = P + (((long)i * H + h) * n + j) * (long)n;
#pragma unroll
    for (int t = 0; t < 8; ++t) prow[t * 32 + lane] = (__bf16)(vals[t] * inv);
}

// APB softmax: one wave per (h,i) row.  S [h,i,j] f32 ; ZB [i,j,h] ; P bf16.
__global__ void softmax_apb_kernel(const float* __restrict__ S, const float* __restrict__ ZB,
                                   __bf16* __restrict__ P, float scale, int n, int H)
{
    const long wid = (long)blockIdx.x * (blockDim.x >> 5) + (threadIdx.x >> 5);
    const int lane = threadIdx.x & 31;
    if (wid >= (long)H * n) return;
    const int h = (int)(wid / n);
    const int i = (int)(wid % n);
    const float* srow = S + ((long)h * n + i) * (long)n;
    float vals[8];
    float mx = -1e30f;
#pragma unroll
    for (int t = 0; t < 8; ++t) {
        const int j = t * 32 + lane;
        const float v = srow[j] * scale + ZB[((long)i * n + j) * H + h];
        vals[t] = v; mx = fmaxf(mx, v);
    }
#pragma unroll
    for (int off = 16; off; off >>= 1) mx = fmaxf(mx, __shfl_xor(mx, off, 32));
    float sum = 0.f;
#pragma unroll
    for (int t = 0; t < 8; ++t) { vals[t] = __expf(vals[t] - mx); sum += vals[t]; }
#pragma unroll
    for (int off = 16; off; off >>= 1) sum += __shfl_xor(sum, off, 32);
    const float inv = 1.0f / sum;
    __bf16* prow = P + ((long)h * n + i) * (long)n;
#pragma unroll
    for (int t = 0; t < 8; ++t) prow[t * 32 + lane] = (__bf16)(vals[t] * inv);
}

// ---------------------------------------------------------------------------
// Host-side helpers
// ---------------------------------------------------------------------------
static inline void gemm(hipStream_t st,
                        const __bf16* A, long la_r, long la_c, long a_b1, long a_b2,
                        const __bf16* B, long lb_r, long lb_c, long b_b1, long b_b2,
                        float* C, long lc_r, long lc_c, long c_b1, long c_b2,
                        const float* bias, __bf16* Cbf,
                        long M, long N, long K, int B1cnt, int B2cnt)
{
    dim3 grid((unsigned)((N + 63) / 64), (unsigned)((M + 63) / 64), (unsigned)(B1cnt * B2cnt));
    gemm_bf16_wmma_kernel<<<grid, 128, 0, st>>>(A, la_r, la_c, a_b1, a_b2,
                                                B, lb_r, lb_c, b_b1, b_b2,
                                                C, lc_r, lc_c, c_b1, c_b2,
                                                bias, Cbf, M, N, K, B2cnt);
}

static inline unsigned eg(long n) { return (unsigned)((n + 255) / 256); }   // elementwise grid
static inline unsigned lg(long r) { return (unsigned)((r + 7) / 8); }       // LN grid (8 waves/blk)

static void run_trimul(hipStream_t st, float* z,
                       const float* lng, const float* lnb, const float* lnog, const float* lnob,
                       const __bf16* wa, const __bf16* wga, const __bf16* wb, const __bf16* wgb,
                       const __bf16* wg, const __bf16* wo, bool outgoing,
                       __bf16* B0, __bf16* B1, __bf16* B2, __bf16* B3,
                       float* F0, float* F1, float* F2, float* F3,
                       __bf16* ATC, __bf16* BTC)
{
    const long R = NPAIR, NC = (long)NTOK * CZ;
    const int swap = outgoing ? 0 : 1;   // incoming stores a[k,i,c] -> At[c][i][k]
    ln_bf16_kernel<<<lg(R), 256, 0, st>>>(z, lng, lnb, B0, R, CZ);
    gemm(st, B0, CZ, 1, 0, 0, wa,  CZ, 1, 0, 0, F1, CZ, 1, 0, 0, nullptr, nullptr, R, CZ, CZ, 1, 1);
    gemm(st, B0, CZ, 1, 0, 0, wga, CZ, 1, 0, 0, F0, CZ, 1, 0, 0, nullptr, nullptr, R, CZ, CZ, 1, 1);
    glu_bf16_kernel<<<eg(R * CZ), 256, 0, st>>>(F0, F1, B1, R * CZ, 0);   // a
    gemm(st, B0, CZ, 1, 0, 0, wb,  CZ, 1, 0, 0, F1, CZ, 1, 0, 0, nullptr, nullptr, R, CZ, CZ, 1, 1);
    gemm(st, B0, CZ, 1, 0, 0, wgb, CZ, 1, 0, 0, F0, CZ, 1, 0, 0, nullptr, nullptr, R, CZ, CZ, 1, 1);
    glu_bf16_kernel<<<eg(R * CZ), 256, 0, st>>>(F0, F1, B2, R * CZ, 0);   // b
    gemm(st, B0, CZ, 1, 0, 0, wg,  CZ, 1, 0, 0, F2, CZ, 1, 0, 0, nullptr, nullptr, R, CZ, CZ, 1, 1);
    // channel-major repack so the 128-batched 256^3 einsum hits unit-stride K
    pair_to_cmajor_bf16_kernel<<<eg(R * CZ), 256, 0, st>>>(B1, ATC, NTOK, CZ, swap);
    pair_to_cmajor_bf16_kernel<<<eg(R * CZ), 256, 0, st>>>(B2, BTC, NTOK, CZ, swap);
    // O[i,j,c] = sum_k At[c][i][k] * Bt[c][j][k]
    gemm(st, ATC, NTOK, 1, (long)NTOK * NTOK, 0,
             BTC, 1, NTOK, (long)NTOK * NTOK, 0,
             F3, NC, CZ, 1, 0,
             nullptr, nullptr, NTOK, NTOK, NTOK, CZ, 1);
    ln_bf16_kernel<<<lg(R), 256, 0, st>>>(F3, lnog, lnob, B3, R, CZ);
    gemm(st, B3, CZ, 1, 0, 0, wo, CZ, 1, 0, 0, F0, CZ, 1, 0, 0, nullptr, nullptr, R, CZ, CZ, 1, 1);
    resgate_kernel<<<eg(R * CZ), 256, 0, st>>>(z, F2, F0, R * CZ);
}

static void run_triattn(hipStream_t st, const float* zin,
                        const float* lng, const float* lnb,
                        const __bf16* wq, const __bf16* wk, const __bf16* wv,
                        const __bf16* wbias, const __bf16* wg, const __bf16* wo,
                        __bf16* B0, __bf16* B1, __bf16* B2, __bf16* B3,
                        float* F0, float* F1, float* F2, float* F3,
                        float* SC, __bf16* PR)
{
    const long R = NPAIR;
    ln_bf16_kernel<<<lg(R), 256, 0, st>>>(zin, lng, lnb, B0, R, CZ);
    gemm(st, B0, CZ, 1, 0, 0, wq, CZ, 1, 0, 0, F0, CZ, 1, 0, 0, nullptr, B1, R, CZ, CZ, 1, 1);
    gemm(st, B0, CZ, 1, 0, 0, wk, CZ, 1, 0, 0, F0, CZ, 1, 0, 0, nullptr, B2, R, CZ, CZ, 1, 1);
    gemm(st, B0, CZ, 1, 0, 0, wv, CZ, 1, 0, 0, F0, CZ, 1, 0, 0, nullptr, B3, R, CZ, CZ, 1, 1);
    gemm(st, B0, CZ, 1, 0, 0, wg, CZ, 1, 0, 0, F1, CZ, 1, 0, 0, nullptr, nullptr, R, CZ, CZ, 1, 1);
    gemm(st, B0, CZ, 1, 0, 0, wbias, HTRI, 1, 0, 0, F2, HTRI, 1, 0, 0, nullptr, nullptr, R, HTRI, CZ, 1, 1);
    // scores[i,h,j,k] = Q[i,j,h,:] . K[i,k,h,:]  (batched over (i,h))
    gemm(st, B1, CZ, 1, (long)NTOK * CZ, DTRI,
             B2, 1, CZ, (long)NTOK * CZ, DTRI,
             SC, NTOK, 1, (long)HTRI * NTOK * NTOK, (long)NTOK * NTOK,
             nullptr, nullptr, NTOK, NTOK, DTRI, NTOK, HTRI);
    softmax_tri_kernel<<<lg((long)NTOK * HTRI * NTOK), 256, 0, st>>>(
        SC, F2, PR, 0.17677669529663687f /*1/sqrt(32)*/, NTOK, HTRI);
    // o[i,j,h,d] = sum_k P * V   -> F3 laid out [n*n, 128]
    gemm(st, PR, NTOK, 1, (long)HTRI * NTOK * NTOK, (long)NTOK * NTOK,
             B3, CZ, 1, (long)NTOK * CZ, DTRI,
             F3, CZ, 1, (long)NTOK * CZ, DTRI,
             nullptr, nullptr, NTOK, DTRI, NTOK, NTOK, HTRI);
    glu_bf16_kernel<<<eg(R * CZ), 256, 0, st>>>(F1, F3, B1, R * CZ, 0);   // sigmoid(g)*o
    gemm(st, B1, CZ, 1, 0, 0, wo, CZ, 1, 0, 0, F0, CZ, 1, 0, 0, nullptr, nullptr, R, CZ, CZ, 1, 1);
    // delta left in F0
}

static void run_transition(hipStream_t st, float* x, long rows, long C, long Hdim,
                           const float* lng, const float* lnb,
                           const __bf16* wa, const __bf16* wb, const __bf16* wo,
                           __bf16* Xbf, float* Af, float* Bf, __bf16* Hbf, float* Delta)
{
    ln_bf16_kernel<<<lg(rows), 256, 0, st>>>(x, lng, lnb, Xbf, rows, C);
    gemm(st, Xbf, C, 1, 0, 0, wa, Hdim, 1, 0, 0, Af, Hdim, 1, 0, 0, nullptr, nullptr, rows, Hdim, C, 1, 1);
    gemm(st, Xbf, C, 1, 0, 0, wb, Hdim, 1, 0, 0, Bf, Hdim, 1, 0, 0, nullptr, nullptr, rows, Hdim, C, 1, 1);
    glu_bf16_kernel<<<eg(rows * Hdim), 256, 0, st>>>(Af, Bf, Hbf, rows * Hdim, 1);  // SiLU
    gemm(st, Hbf, Hdim, 1, 0, 0, wo, C, 1, 0, 0, Delta, C, 1, 0, 0, nullptr, nullptr, rows, C, Hdim, 1, 1);
    add_inplace_kernel<<<eg(rows * C), 256, 0, st>>>(x, Delta, rows * C);
}

static void run_apb(hipStream_t st, float* s, const float* z,
                    const float* bq, const float* lnag, const float* lnab,
                    const float* lnzg, const float* lnzb,
                    const __bf16* wbias, const __bf16* wq, const __bf16* wk,
                    const __bf16* wv, const __bf16* wg, const __bf16* wo,
                    __bf16* B0, __bf16* B1, __bf16* B2,
                    float* F0, float* F1, float* F2, float* F3)
{
    const long R = NPAIR;
    // zb = LN(z) @ w_bias -> F0 [65536,16]
    ln_bf16_kernel<<<lg(R), 256, 0, st>>>(z, lnzg, lnzb, B0, R, CZ);
    gemm(st, B0, CZ, 1, 0, 0, wbias, HS, 1, 0, 0, F0, HS, 1, 0, 0, nullptr, nullptr, R, HS, CZ, 1, 1);
    // a = LN(s) -> B2
    ln_bf16_kernel<<<lg(NTOK), 256, 0, st>>>(s, lnag, lnab, B2, NTOK, CS);
    __bf16* qbf = B0;                       // z-LN no longer needed: carve slices
    __bf16* kbf = B0 + (long)NTOK * CS;
    __bf16* vbf = B0 + 2L * NTOK * CS;
    gemm(st, B2, CS, 1, 0, 0, wq, CS, 1, 0, 0, F1, CS, 1, 0, 0, bq,      qbf, NTOK, CS, CS, 1, 1);
    gemm(st, B2, CS, 1, 0, 0, wk, CS, 1, 0, 0, F1, CS, 1, 0, 0, nullptr, kbf, NTOK, CS, CS, 1, 1);
    gemm(st, B2, CS, 1, 0, 0, wv, CS, 1, 0, 0, F1, CS, 1, 0, 0, nullptr, vbf, NTOK, CS, CS, 1, 1);
    gemm(st, B2, CS, 1, 0, 0, wg, CS, 1, 0, 0, F2, CS, 1, 0, 0, nullptr, nullptr, NTOK, CS, CS, 1, 1);
    // scores[h,i,j] (batched over h, K=24 zero-padded in the GEMM)
    gemm(st, qbf, CS, 1, D_S, 0,  kbf, 1, CS, D_S, 0,
             F3, NTOK, 1, (long)NTOK * NTOK, 0,
             nullptr, nullptr, NTOK, NTOK, D_S, HS, 1);
    softmax_apb_kernel<<<lg((long)HS * NTOK), 256, 0, st>>>(
        F3, F0, B1, 0.20412414523193154f /*1/sqrt(24)*/, NTOK, HS);
    // o[i, h*24+d]
    gemm(st, B1, NTOK, 1, (long)NTOK * NTOK, 0,
             vbf, CS, 1, D_S, 0,
             F1, CS, 1, D_S, 0,
             nullptr, nullptr, NTOK, D_S, NTOK, HS, 1);
    glu_bf16_kernel<<<eg((long)NTOK * CS), 256, 0, st>>>(F2, F1, B2, (long)NTOK * CS, 0);
    gemm(st, B2, CS, 1, 0, 0, wo, CS, 1, 0, 0, F0, CS, 1, 0, 0, nullptr, nullptr, NTOK, CS, CS, 1, 1);
    add_inplace_kernel<<<eg((long)NTOK * CS), 256, 0, st>>>(s, F0, (long)NTOK * CS);
}

// Weight table: input index assumes jax pytree (alphabetical) flatten order:
// params{apb, ta_end, ta_start, tm_in, tm_out, trans_s, trans_z}, s, z.
enum {
    W_TMO_A, W_TMO_GA, W_TMO_B, W_TMO_GB, W_TMO_G, W_TMO_O,
    W_TMI_A, W_TMI_GA, W_TMI_B, W_TMI_GB, W_TMI_G, W_TMI_O,
    W_TAS_Q, W_TAS_K, W_TAS_V, W_TAS_B, W_TAS_G, W_TAS_O,
    W_TAE_Q, W_TAE_K, W_TAE_V, W_TAE_B, W_TAE_G, W_TAE_O,
    W_TRZ_A, W_TRZ_B, W_TRZ_O,
    W_APB_BIAS, W_APB_Q, W_APB_K, W_APB_V, W_APB_G, W_APB_O,
    W_TRS_A, W_TRS_B, W_TRS_O,
    NW
};
static const int  widx[NW] = {
    41, 44, 42, 45, 43, 46,
    31, 34, 32, 35, 33, 36,
    25, 23, 26, 21, 22, 24,
    17, 15, 18, 13, 14, 16,
    54, 55, 56,
    5, 9, 7, 10, 6, 8,
    49, 50, 51
};
static const long welems[NW] = {
    16384, 16384, 16384, 16384, 16384, 16384,
    16384, 16384, 16384, 16384, 16384, 16384,
    16384, 16384, 16384, 512,   16384, 16384,
    16384, 16384, 16384, 512,   16384, 16384,
    65536, 65536, 65536,
    2048, 147456, 147456, 147456, 147456, 147456,
    589824, 589824, 589824
};

extern "C" void kernel_launch(void* const* d_in, const int* in_sizes, int n_in,
                              void* d_out, int out_size, void* d_ws, size_t ws_size,
                              hipStream_t stream)
{
    (void)in_sizes; (void)n_in; (void)out_size; (void)ws_size;
    const float* in[64];
    for (int i = 0; i < 64 && i < n_in; ++i) in[i] = (const float*)d_in[i];

    char* wsp = (char*)d_ws;
    size_t used = 0;
    auto alloc = [&](size_t bytes) -> void* {
        void* r = wsp + used;
        used += (bytes + 255) & ~(size_t)255;
        return r;
    };

    // bf16 weight arena (~6 MB)
    __bf16* wbf[NW];
    for (int i = 0; i < NW; ++i) {
        wbf[i] = (__bf16*)alloc((size_t)welems[i] * sizeof(__bf16));
        f32_to_bf16_kernel<<<eg(welems[i]), 256, 0, stream>>>(in[widx[i]], wbf[i], welems[i]);
    }

    // activation scratch
    __bf16* B0 = (__bf16*)alloc((size_t)NPAIR * CZ * sizeof(__bf16));
    __bf16* B1 = (__bf16*)alloc((size_t)NPAIR * CZ * sizeof(__bf16));
    __bf16* B2 = (__bf16*)alloc((size_t)NPAIR * CZ * sizeof(__bf16));
    __bf16* B3 = (__bf16*)alloc((size_t)NPAIR * CZ * sizeof(__bf16));
    float*  F0 = (float*)alloc((size_t)NPAIR * CZ * sizeof(float));
    float*  F1 = (float*)alloc((size_t)NPAIR * CZ * sizeof(float));
    float*  F2 = (float*)alloc((size_t)NPAIR * CZ * sizeof(float));
    float*  F3 = (float*)alloc((size_t)NPAIR * CZ * sizeof(float));
    float*  SC = (float*)alloc((size_t)NPAIR * HTRI * NTOK * sizeof(float));    // 268 MB
    __bf16* PR = (__bf16*)alloc((size_t)NPAIR * HTRI * NTOK * sizeof(__bf16));  // 134 MB
    float*  ZT = (float*)alloc((size_t)NPAIR * CZ * sizeof(float));
    // trimul channel-major scratch aliases the (idle there) scores arena
    __bf16* ATC = (__bf16*)SC;
    __bf16* BTC = ATC + (size_t)NPAIR * CZ;

    float* s_out = (float*)d_out;
    float* z_out = s_out + (long)NTOK * CS;
    hipMemcpyAsync(s_out, in[57], (size_t)NTOK * CS * sizeof(float),
                   hipMemcpyDeviceToDevice, stream);
    hipMemcpyAsync(z_out, in[58], (size_t)NPAIR * CZ * sizeof(float),
                   hipMemcpyDeviceToDevice, stream);

    // z = z + trimul(z, tm_out, outgoing)
    run_trimul(stream, z_out, in[38], in[37], in[40], in[39],
               wbf[W_TMO_A], wbf[W_TMO_GA], wbf[W_TMO_B], wbf[W_TMO_GB],
               wbf[W_TMO_G], wbf[W_TMO_O], true,
               B0, B1, B2, B3, F0, F1, F2, F3, ATC, BTC);
    // z = z + trimul(z, tm_in, incoming)
    run_trimul(stream, z_out, in[28], in[27], in[30], in[29],
               wbf[W_TMI_A], wbf[W_TMI_GA], wbf[W_TMI_B], wbf[W_TMI_GB],
               wbf[W_TMI_G], wbf[W_TMI_O], false,
               B0, B1, B2, B3, F0, F1, F2, F3, ATC, BTC);
    // z = z + triattn(z, ta_start)
    run_triattn(stream, z_out, in[20], in[19],
                wbf[W_TAS_Q], wbf[W_TAS_K], wbf[W_TAS_V],
                wbf[W_TAS_B], wbf[W_TAS_G], wbf[W_TAS_O],
                B0, B1, B2, B3, F0, F1, F2, F3, SC, PR);
    add_inplace_kernel<<<eg((long)NPAIR * CZ), 256, 0, stream>>>(z_out, F0, (long)NPAIR * CZ);
    // z = z + triattn(z^T, ta_end)^T
    transpose_pair_kernel<<<eg((long)NPAIR * CZ), 256, 0, stream>>>(z_out, ZT, NTOK, CZ);
    run_triattn(stream, ZT, in[12], in[11],
                wbf[W_TAE_Q], wbf[W_TAE_K], wbf[W_TAE_V],
                wbf[W_TAE_B], wbf[W_TAE_G], wbf[W_TAE_O],
                B0, B1, B2, B3, F0, F1, F2, F3, SC, PR);
    add_transposed_kernel<<<eg((long)NPAIR * CZ), 256, 0, stream>>>(z_out, F0, NTOK, CZ);
    // z = z + transition(z)  (hidden [65536,512] lives in the SC/PR arenas)
    run_transition(stream, z_out, NPAIR, CZ, 4 * CZ, in[53], in[52],
                   wbf[W_TRZ_A], wbf[W_TRZ_B], wbf[W_TRZ_O],
                   B0, SC, SC + (size_t)NPAIR * (4 * CZ), PR, F2);
    // s = s + apb(s, z)
    run_apb(stream, s_out, z_out, in[0], in[2], in[1], in[4], in[3],
            wbf[W_APB_BIAS], wbf[W_APB_Q], wbf[W_APB_K], wbf[W_APB_V],
            wbf[W_APB_G], wbf[W_APB_O],
            B0, B1, B2, F0, F1, F2, F3);
    // s = s + transition(s)
    run_transition(stream, s_out, NTOK, CS, 4 * CS, in[48], in[47],
                   wbf[W_TRS_A], wbf[W_TRS_B], wbf[W_TRS_O],
                   B0, F0, F1, B1, F2);
}
